// SDFNetwork_55430847922694
// MI455X (gfx1250) — matrix operand order repaired and checked by reading, not compiled, not run on hardware
//
#include <hip/hip_runtime.h>
#include <hip/hip_bf16.h>

typedef __attribute__((ext_vector_type(16))) _Float16 v16h;
typedef __attribute__((ext_vector_type(8)))  float    v8f;

#define NLEVELS 16
#define TSIZE   (1u << 19)   // hashmap entries per level
#define HIDDEN  64
#define PTS_PER_BLOCK 128    // 8 waves x 16 points

// floor(16 * 1.3819^l) for l = 0..15 (precomputed, matches reference _level_res)
__device__ constexpr int kResC[16] = {
    16, 22, 30, 42, 58, 80, 111, 153, 212, 294, 406, 561, 775, 1072, 1481, 2047};
// dense iff (res+1)^3 <= 2^19  -> levels 0..4 dense, 5..15 hashed
__device__ constexpr bool kDense[16] = {
    true, true, true, true, true, false, false, false,
    false, false, false, false, false, false, false, false};

__device__ inline v16h load_b16(const _Float16* __restrict__ p) {
  v16h r;
#pragma unroll
  for (int i = 0; i < 16; ++i) r[i] = p[i];
  return r;
}

// Build A-operand vector for v_wmma_f32_16x16x32_f16 from a 32/64-half LDS row.
// Halves 0..7 <- row[base .. base+7], halves 8..15 <- row[base+16 .. base+23].
__device__ inline v16h load_a16(const _Float16* __restrict__ row, int base) {
  v16h r;
#pragma unroll
  for (int i = 0; i < 8; ++i) {
    r[i]     = row[base + i];
    r[i + 8] = row[base + 16 + i];
  }
  return r;
}

// Encode 8 consecutive levels [LBASE, LBASE+8) of the hashgrid for one point.
template <int LBASE>
__device__ inline void encode8(const float* __restrict__ tables,
                               float px, float py, float pz,
                               _Float16* __restrict__ encRow) {
#pragma unroll
  for (int ll = 0; ll < 8; ++ll) {
    const int   l   = LBASE + ll;
    const int   res = kResC[l];
    const float2* tb = (const float2*)(tables + (size_t)l * (size_t)TSIZE * 2u);

    const float fx = px * (float)res;
    const float fy = py * (float)res;
    const float fz = pz * (float)res;
    const float bx = floorf(fx), by = floorf(fy), bz = floorf(fz);
    const float rx = fx - bx,   ry = fy - by,   rz = fz - bz;
    const unsigned X = (unsigned)bx, Y = (unsigned)by, Z = (unsigned)bz;

    float a0 = 0.0f, a1 = 0.0f;
#pragma unroll
    for (int o = 0; o < 8; ++o) {
      const unsigned cx = X + ((o >> 2) & 1);
      const unsigned cy = Y + ((o >> 1) & 1);
      const unsigned cz = Z + (o & 1);
      const float w = ((o & 4) ? rx : 1.0f - rx) *
                      ((o & 2) ? ry : 1.0f - ry) *
                      ((o & 1) ? rz : 1.0f - rz);
      unsigned idx;
      if (kDense[l]) {
        const unsigned r1 = (unsigned)(res + 1);
        idx = cx + cy * r1 + cz * r1 * r1;
      } else {
        idx = cx ^ (cy * 2654435761u) ^ (cz * 805459861u);
        idx &= (TSIZE - 1u);
      }
      const float2 f = tb[idx];          // L2-resident gather (tables = 64MB < 192MB L2)
      a0 = fmaf(w, f.x, a0);
      a1 = fmaf(w, f.y, a1);
    }
    encRow[2 * l]     = (_Float16)a0;
    encRow[2 * l + 1] = (_Float16)a1;
  }
}

__global__ __launch_bounds__(256)
void sdf_fused_kernel(const float* __restrict__ xin,     // [N,3]
                      const float* __restrict__ tables,  // [16, 2^19, 2]
                      const float* __restrict__ W0,      // [32,64]
                      const float* __restrict__ W1,      // [64,64]
                      const float* __restrict__ W2,      // [64,1]
                      float* __restrict__ out,           // [N]
                      unsigned n) {
  // B operand layout (K x 16 f16): lanes 0..15 hold K=0..7,16..23 of col N=lane,
  // lanes 16..31 hold K=8..15,24..31 of col N=lane-16 (matches load_a16 order).
  __shared__ __align__(16) _Float16 sW0[4][32][16];      //  4 KB: W0, 4 N-blocks
  __shared__ __align__(16) _Float16 sW1[4][2][32][16];   //  8 KB: W1, 4 N-blk x 2 K-blk
  __shared__ float                   sW2[64];            // 256 B
  __shared__ __align__(16) _Float16 sEnc[PTS_PER_BLOCK][32];  // 8 KB encodings
  __shared__ __align__(16) _Float16 sH[8][16][64];       // 16 KB hidden acts (per wave)

  const int tid  = threadIdx.x;
  const int wave = tid >> 5;
  const int lane = tid & 31;

  // ---- Phase 0: stage weights into swizzled B-operand layout -------------
  for (int t = tid; t < 4 * 32 * 16; t += 256) {
    const int nb = t >> 9, rem = t & 511, ln = rem >> 4, i = rem & 15;
    const int k = 8 * (ln >> 4) + i + ((i >= 8) ? 8 : 0);
    const int nn = nb * 16 + (ln & 15);
    sW0[nb][ln][i] = (_Float16)W0[k * HIDDEN + nn];
  }
  for (int t = tid; t < 4 * 2 * 32 * 16; t += 256) {
    const int nb = t >> 10, kb = (t >> 9) & 1, ln = (t >> 4) & 31, i = t & 15;
    const int k = kb * 32 + 8 * (ln >> 4) + i + ((i >= 8) ? 8 : 0);
    const int nn = nb * 16 + (ln & 15);
    sW1[nb][kb][ln][i] = (_Float16)W1[k * HIDDEN + nn];
  }
  if (tid < 64) sW2[tid] = W2[tid];

  // ---- Phase 1: hashgrid encode (2 lanes per point, 8 levels each) -------
  {
    const int pw   = lane >> 1;       // point within wave 0..15
    const int hsel = lane & 1;        // level half
    const int pblk = wave * 16 + pw;  // point within block 0..127
    unsigned gp = blockIdx.x * (unsigned)PTS_PER_BLOCK + (unsigned)pblk;
    const unsigned gpl = (gp < n) ? gp : (n - 1u);  // clamp; keep EXEC full
    const float x0 = (xin[3u * gpl + 0u] + 1.0f) * 0.5f;
    const float x1 = (xin[3u * gpl + 1u] + 1.0f) * 0.5f;
    const float x2 = (xin[3u * gpl + 2u] + 1.0f) * 0.5f;
    _Float16* encRow = &sEnc[pblk][0];
    if (hsel == 0) encode8<0>(tables, x0, x1, x2, encRow);
    else           encode8<8>(tables, x0, x1, x2, encRow);
  }
  __syncthreads();

  // ---- Phase 2: fused MLP via WMMA (one 16-point M-tile per wave) --------
  const int ncol = lane & 15;          // column within C/D block
  const int mhi  = (lane >> 4) << 3;   // row offset: lanes 16..31 hold M=8..15

  // Layer 0: enc[16,32] @ W0[32,64]
  const _Float16* aRow = &sEnc[wave * 16 + ncol][0];
  const int abase = (lane < 16) ? 0 : 8;
  const v16h a = load_a16(aRow, abase);
  v8f d0, d1, d2, d3;
  {
    v8f c0 = {}, c1 = {}, c2 = {}, c3 = {};
    d0 = __builtin_amdgcn_wmma_f32_16x16x32_f16(false, a, false, load_b16(&sW0[0][lane][0]), (short)0, c0, false, false);
    d1 = __builtin_amdgcn_wmma_f32_16x16x32_f16(false, a, false, load_b16(&sW0[1][lane][0]), (short)0, c1, false, false);
    d2 = __builtin_amdgcn_wmma_f32_16x16x32_f16(false, a, false, load_b16(&sW0[2][lane][0]), (short)0, c2, false, false);
    d3 = __builtin_amdgcn_wmma_f32_16x16x32_f16(false, a, false, load_b16(&sW0[3][lane][0]), (short)0, c3, false, false);
  }
  // ReLU -> f16 -> LDS, using documented C/D layout (lane=col, VGPR r=row)
  {
    _Float16(*hT)[64] = sH[wave];
#pragma unroll
    for (int r = 0; r < 8; ++r) {
      hT[mhi + r][0 * 16 + ncol] = (_Float16)fmaxf(d0[r], 0.0f);
      hT[mhi + r][1 * 16 + ncol] = (_Float16)fmaxf(d1[r], 0.0f);
      hT[mhi + r][2 * 16 + ncol] = (_Float16)fmaxf(d2[r], 0.0f);
      hT[mhi + r][3 * 16 + ncol] = (_Float16)fmaxf(d3[r], 0.0f);
    }
  }
  __syncthreads();

  // Layer 1: h[16,64] @ W1[64,64]  (K split into 2 blocks, accumulate via C)
  const _Float16* hRow = &sH[wave][ncol][0];
  const v16h h0 = load_a16(hRow, abase);        // K 0..31
  const v16h h1 = load_a16(hRow, 32 + abase);   // K 32..63
  v8f e[4];
#pragma unroll
  for (int nb = 0; nb < 4; ++nb) {
    v8f c = {};
    c = __builtin_amdgcn_wmma_f32_16x16x32_f16(false, h0, false, load_b16(&sW1[nb][0][lane][0]), (short)0, c, false, false);
    c = __builtin_amdgcn_wmma_f32_16x16x32_f16(false, h1, false, load_b16(&sW1[nb][1][lane][0]), (short)0, c, false, false);
    e[nb] = c;
  }

  // Layer 2: relu(h2)[16,64] @ W2[64,1] — per-lane partial dot + xor-tree
  float w2v0 = sW2[0 * 16 + ncol];
  float w2v1 = sW2[1 * 16 + ncol];
  float w2v2 = sW2[2 * 16 + ncol];
  float w2v3 = sW2[3 * 16 + ncol];
  const unsigned gbase = blockIdx.x * (unsigned)PTS_PER_BLOCK + (unsigned)(wave * 16);
#pragma unroll
  for (int r = 0; r < 8; ++r) {
    float s = fmaxf(e[0][r], 0.0f) * w2v0 + fmaxf(e[1][r], 0.0f) * w2v1 +
              fmaxf(e[2][r], 0.0f) * w2v2 + fmaxf(e[3][r], 0.0f) * w2v3;
    s += __shfl_xor(s, 1, 32);
    s += __shfl_xor(s, 2, 32);
    s += __shfl_xor(s, 4, 32);
    s += __shfl_xor(s, 8, 32);   // masks <=8: reduction stays inside each 16-lane half
    const unsigned gout = gbase + (unsigned)(mhi + r);
    if (ncol == 0 && gout < n) out[gout] = s;
  }
}

extern "C" void kernel_launch(void* const* d_in, const int* in_sizes, int n_in,
                              void* d_out, int out_size, void* d_ws, size_t ws_size,
                              hipStream_t stream) {
  const float* x  = (const float*)d_in[0];
  const float* tb = (const float*)d_in[1];
  const float* W0 = (const float*)d_in[2];
  const float* W1 = (const float*)d_in[3];
  const float* W2 = (const float*)d_in[4];
  float* out = (float*)d_out;
  const unsigned npts = (unsigned)(in_sizes[0] / 3);
  const unsigned blocks = (npts + PTS_PER_BLOCK - 1) / PTS_PER_BLOCK;
  sdf_fused_kernel<<<blocks, 256, 0, stream>>>(x, tb, W0, W1, W2, out, npts);
}
